// ContrastiveSamplingNetwork_4002909520152
// MI455X (gfx1250) — matrix operand order repaired and checked by reading, not compiled, not run on hardware
//
#include <hip/hip_runtime.h>
#include <stdint.h>

// ---------------------------------------------------------------------------
// Types for CDNA5 WMMA (wave32). 16x16x32 bf16 -> f32 accumulate.
// ---------------------------------------------------------------------------
typedef __attribute__((ext_vector_type(16))) __bf16 v16bf;
typedef __attribute__((ext_vector_type(8)))  float  v8f;
typedef __attribute__((ext_vector_type(4)))  int    v4i_t;

#define NTOT      16384
#define INV_NTOT  (1.0f/16384.0f)
#define NPTS      4096
#define NB        4
#define INV_TEMP  10.0f
#define POS_T     0.7f
#define NEG_T     0.3f
#define MARGIN    0.2f

// Async global->LDS (ASYNCcnt-tracked DMA-style copy), guarded so the file
// still compiles on toolchains without the gfx1250 async builtins.
#if defined(__has_builtin)
#  if __has_builtin(__builtin_amdgcn_global_load_async_to_lds_b128)
#    define USE_ASYNC_LDS 1
#  endif
#endif
#ifndef USE_ASYNC_LDS
#  define USE_ASYNC_LDS 0
#endif

__device__ __forceinline__ void async_wait0() {
#if USE_ASYNC_LDS
#  if __has_builtin(__builtin_amdgcn_s_wait_asynccnt)
  __builtin_amdgcn_s_wait_asynccnt(0);
#  else
  asm volatile("s_wait_asynccnt 0x0" ::: "memory");
#  endif
#endif
}

// Stage a 128x64 bf16 panel (16 KB = 1024 x b128) with 256 threads.
__device__ __forceinline__ void stage_panel(const unsigned short* __restrict__ src,
                                            unsigned short* dst, int tid) {
#if USE_ASYNC_LDS
#pragma unroll
  for (int i = 0; i < 4; ++i) {
    const unsigned short* s = src + (size_t)(tid + i * 256) * 8;
    unsigned short*       d = dst + (tid + i * 256) * 8;
    __builtin_amdgcn_global_load_async_to_lds_b128(
        (__attribute__((address_space(1))) v4i_t*)s,
        (__attribute__((address_space(3))) v4i_t*)d, 0, 0);
  }
#else
#pragma unroll
  for (int i = 0; i < 4; ++i)
    ((uint4*)dst)[tid + i * 256] = ((const uint4*)src)[tid + i * 256];
  __builtin_prefetch(src + 128 * 64, 0, 0);
#endif
}

struct BFrag32B { uint4 lo; uint4 hi; };   // 32 bytes == 16 bf16

__device__ __forceinline__ v16bf load_frag(const unsigned short* lds, int e0, int e1) {
  BFrag32B f;
  f.lo = *reinterpret_cast<const uint4*>(lds + e0);   // 8 bf16
  f.hi = *reinterpret_cast<const uint4*>(lds + e1);   // 8 bf16
  return __builtin_bit_cast(v16bf, f);
}

__device__ __forceinline__ float wave_sum(float v) {
#pragma unroll
  for (int m = 16; m >= 1; m >>= 1) v += __shfl_xor(v, m, 32);  // wave32
  return v;
}

__device__ __forceinline__ float bn_relu(float x, float sum, float sq, float g, float be) {
  float mu  = sum * INV_NTOT;
  float var = sq  * INV_NTOT - mu * mu;
  float y = (x - mu) * rsqrtf(var + 1e-5f) * g + be;
  return fmaxf(y, 0.0f);
}

__device__ __forceinline__ unsigned short f32_to_bf16(float x) {
  unsigned int u = __builtin_bit_cast(unsigned int, x);
  u += 0x7FFFu + ((u >> 16) & 1u);   // round-to-nearest-even
  return (unsigned short)(u >> 16);
}

// ---------------------------------------------------------------------------
// Workspace zeroing (BN stats + loss accumulators)
// ---------------------------------------------------------------------------
__global__ void k_zero(float* __restrict__ p, int n) {
  int i = blockIdx.x * blockDim.x + threadIdx.x;
  if (i < n) p[i] = 0.0f;
}

// ---------------------------------------------------------------------------
// Feature extractor layer 1: xyz(3) -> 32, collect BN stats
// ---------------------------------------------------------------------------
__global__ __launch_bounds__(256) void k_fe1(const float* __restrict__ xyz,
    const float* __restrict__ w1, const float* __restrict__ b1,
    float* __restrict__ a1, float* __restrict__ st) {
  int p = blockIdx.x * 256 + threadIdx.x;
  int lane = threadIdx.x & 31;
  float x = xyz[p * 3 + 0], y = xyz[p * 3 + 1], z = xyz[p * 3 + 2];
#pragma unroll
  for (int c = 0; c < 32; ++c) {
    float v = fmaf(x, w1[c], fmaf(y, w1[32 + c], fmaf(z, w1[64 + c], b1[c])));
    a1[p * 32 + c] = v;
    float s = wave_sum(v);
    float q = wave_sum(v * v);
    if (lane == 0) { atomicAdd(&st[c], s); atomicAdd(&st[32 + c], q); }
  }
}

// ---------------------------------------------------------------------------
// FE layer 2: bn+relu(32) -> 64, collect BN stats
// ---------------------------------------------------------------------------
__global__ __launch_bounds__(256) void k_fe2(const float* __restrict__ a1,
    const float* __restrict__ st1, const float* __restrict__ g1, const float* __restrict__ be1,
    const float* __restrict__ w2, const float* __restrict__ b2,
    float* __restrict__ a2, float* __restrict__ st2) {
  int p = blockIdx.x * 256 + threadIdx.x;
  int lane = threadIdx.x & 31;
  float h[32];
#pragma unroll
  for (int c = 0; c < 32; ++c)
    h[c] = bn_relu(a1[p * 32 + c], st1[c], st1[32 + c], g1[c], be1[c]);
  for (int o = 0; o < 64; ++o) {
    float acc = b2[o];
#pragma unroll
    for (int c = 0; c < 32; ++c) acc = fmaf(h[c], w2[c * 64 + o], acc);
    a2[p * 64 + o] = acc;
    float s = wave_sum(acc), q = wave_sum(acc * acc);
    if (lane == 0) { atomicAdd(&st2[o], s); atomicAdd(&st2[64 + o], q); }
  }
}

// ---------------------------------------------------------------------------
// FE layer 3: bn+relu(64) -> feat_raw(64)  (written to d_out feat region)
// ---------------------------------------------------------------------------
__global__ __launch_bounds__(256) void k_fe3(const float* __restrict__ a2,
    const float* __restrict__ st2, const float* __restrict__ g2, const float* __restrict__ be2,
    const float* __restrict__ w3, const float* __restrict__ b3,
    float* __restrict__ fraw) {
  int p = blockIdx.x * 256 + threadIdx.x;
  float h[64];
#pragma unroll
  for (int c = 0; c < 64; ++c)
    h[c] = bn_relu(a2[p * 64 + c], st2[c], st2[64 + c], g2[c], be2[c]);
  for (int d = 0; d < 64; ++d) {
    float acc = b3[d];
#pragma unroll
    for (int c = 0; c < 64; ++c) acc = fmaf(h[c], w3[c * 64 + d], acc);
    fraw[p * 64 + d] = acc;
  }
}

// ---------------------------------------------------------------------------
// Normalize feat (in place), emit bf16 copy, score-net layer 1 (67 -> 128)
// ---------------------------------------------------------------------------
__global__ __launch_bounds__(256) void k_sn1(float* __restrict__ feat,
    unsigned short* __restrict__ featbf, const float* __restrict__ xyz,
    const float* __restrict__ w1, const float* __restrict__ b1,
    float* __restrict__ s1, float* __restrict__ st) {
  int p = blockIdx.x * 256 + threadIdx.x;
  int lane = threadIdx.x & 31;
  float f[64]; float n2 = 0.0f;
#pragma unroll
  for (int d = 0; d < 64; ++d) { f[d] = feat[p * 64 + d]; n2 = fmaf(f[d], f[d], n2); }
  float inv = 1.0f / fmaxf(sqrtf(n2), 1e-12f);
#pragma unroll
  for (int d = 0; d < 64; ++d) {
    f[d] *= inv;
    feat[p * 64 + d]   = f[d];
    featbf[p * 64 + d] = f32_to_bf16(f[d]);
  }
  float x = xyz[p * 3 + 0], y = xyz[p * 3 + 1], z = xyz[p * 3 + 2];
  for (int o = 0; o < 128; ++o) {
    float acc = b1[o];
    acc = fmaf(x, w1[o], acc);
    acc = fmaf(y, w1[128 + o], acc);
    acc = fmaf(z, w1[256 + o], acc);
#pragma unroll
    for (int c = 0; c < 64; ++c) acc = fmaf(f[c], w1[(3 + c) * 128 + o], acc);
    s1[p * 128 + o] = acc;
    float s = wave_sum(acc), q = wave_sum(acc * acc);
    if (lane == 0) { atomicAdd(&st[o], s); atomicAdd(&st[128 + o], q); }
  }
}

// ---------------------------------------------------------------------------
// Score-net layer 2: bn+relu(128) -> 64
// ---------------------------------------------------------------------------
__global__ __launch_bounds__(256) void k_sn2(const float* __restrict__ s1,
    const float* __restrict__ st1, const float* __restrict__ g1, const float* __restrict__ be1,
    const float* __restrict__ w2, const float* __restrict__ b2,
    float* __restrict__ s2, float* __restrict__ st2) {
  int p = blockIdx.x * 256 + threadIdx.x;
  int lane = threadIdx.x & 31;
  float h[128];
#pragma unroll
  for (int c = 0; c < 128; ++c)
    h[c] = bn_relu(s1[p * 128 + c], st1[c], st1[128 + c], g1[c], be1[c]);
  for (int o = 0; o < 64; ++o) {
    float acc = b2[o];
#pragma unroll
    for (int c = 0; c < 128; ++c) acc = fmaf(h[c], w2[c * 64 + o], acc);
    s2[p * 64 + o] = acc;
    float s = wave_sum(acc), q = wave_sum(acc * acc);
    if (lane == 0) { atomicAdd(&st2[o], s); atomicAdd(&st2[64 + o], q); }
  }
}

// ---------------------------------------------------------------------------
// Score-net layer 3: bn+relu(64) -> sigmoid scalar prob
// ---------------------------------------------------------------------------
__global__ __launch_bounds__(256) void k_sn3(const float* __restrict__ s2,
    const float* __restrict__ st2, const float* __restrict__ g2, const float* __restrict__ be2,
    const float* __restrict__ w3, const float* __restrict__ b3,
    float* __restrict__ probs) {
  int p = blockIdx.x * 256 + threadIdx.x;
  float logit = b3[0];
#pragma unroll
  for (int c = 0; c < 64; ++c) {
    float h = bn_relu(s2[p * 64 + c], st2[c], st2[64 + c], g2[c], be2[c]);
    logit = fmaf(h, w3[c], logit);
  }
  probs[p] = 1.0f / (1.0f + expf(-logit));
}

// ---------------------------------------------------------------------------
// Fused: sim = feat@feat^T / TEMP via bf16 WMMA with double-buffered
// async global->LDS panel staging; contrastive loss sums; per-row top-17.
// One workgroup per 16-row strip per batch; 8 wave32s; 128-column panels.
// ---------------------------------------------------------------------------
__global__ __launch_bounds__(256) void k_sim(const unsigned short* __restrict__ featbf,
                                             const float* __restrict__ probs,
                                             int* __restrict__ knn,
                                             float* __restrict__ lacc) {
  __shared__ unsigned short lds_rows[16 * 64];        // 2 KB  : A strip (bf16)
  __shared__ unsigned short lds_cols[2][128 * 64];    // 2x16KB: B panels (bf16)
  __shared__ float lds_sim[128 * 16];                 // 8 KB  : sim, col-major
  __shared__ float lds_prow[16];
  __shared__ float lds_red[32];

  const int bb   = blockIdx.y;
  const int r0   = blockIdx.x * 16;
  const int tid  = threadIdx.x;
  const int lane = tid & 31;
  const int wv   = tid >> 5;

  const unsigned short* fb = featbf + (size_t)bb * NPTS * 64;
  const float* pb = probs + bb * NPTS;

  if (tid < 128) ((uint4*)lds_rows)[tid] = ((const uint4*)(fb + (size_t)r0 * 64))[tid];
  if (tid < 16)  lds_prow[tid] = pb[r0 + tid];
  stage_panel(fb, lds_cols[0], tid);     // panel 0 (async if available)
  async_wait0();
  __syncthreads();

  // A fragments: ISA 16-bit A 16x32 layout: lane<16 holds K{0..7,16..23} of
  // row (lane&15); lane>=16 holds K{8..15,24..31}.
  const int arow  = lane & 15;
  const int ahalf = (lane < 16) ? 0 : 8;
  const v16bf afrag0 = load_frag(lds_rows, arow * 64 +  0 + ahalf, arow * 64 + 16 + ahalf);
  const v16bf afrag1 = load_frag(lds_rows, arow * 64 + 32 + ahalf, arow * 64 + 48 + ahalf);

  const int mbase = (lane < 16) ? 0 : 8;   // C layout: lane half selects M block
  const int bcol  = lane & 15;             // B layout: lane&15 = column N
  const int bkh   = (lane < 16) ? 0 : 8;   // lane half selects K half
  const int c0    = wv * 16;               // this wave's column tile offset

  float pr[8];                             // row probs for this lane's M's
#pragma unroll
  for (int r = 0; r < 8; ++r) pr[r] = lds_prow[mbase + r];

  float posd = 0.f, posc = 0.f, negd = 0.f, negc = 0.f;
  float tv[17]; int ti[17];
#pragma unroll
  for (int s = 0; s < 17; ++s) { tv[s] = -3.0e38f; ti[s] = 0; }

  int cur = 0;
  for (int j0 = 0; j0 < NPTS; j0 += 128) {
    // Kick off next panel while computing this one (prev readers of buf^1
    // finished before last barrier).
    if (j0 + 128 < NPTS)
      stage_panel(fb + (size_t)(j0 + 128) * 64, lds_cols[cur ^ 1], tid);

    const unsigned short* cbuf = lds_cols[cur];
    const int bbase = (c0 + bcol) * 64;
    v16bf bf0 = load_frag(cbuf, bbase +  0 + bkh, bbase + 16 + bkh);
    v16bf bf1 = load_frag(cbuf, bbase + 32 + bkh, bbase + 48 + bkh);

    v8f acc = {};
    acc = __builtin_amdgcn_wmma_f32_16x16x32_bf16(false, afrag0, false, bf0,
                                                  (short)0, acc, false, false);
    acc = __builtin_amdgcn_wmma_f32_16x16x32_bf16(false, afrag1, false, bf1,
                                                  (short)0, acc, false, false);

    const float pc = pb[j0 + c0 + bcol];
    float sv[8];
#pragma unroll
    for (int r = 0; r < 8; ++r) {
      float sim = acc[r] * INV_TEMP;
      sv[r] = sim;
      float pd = fabsf(pr[r] - pc);
      if (sim > POS_T) { posd += pd; posc += 1.0f; }
      if (sim < NEG_T) { negd += fmaxf(MARGIN - pd, 0.0f); negc += 1.0f; }
    }
    // Column-major sim strip: this lane's 8 M-values are contiguous
    // -> two ds_store_b128 instead of 8 scalar stores.
    float4* dst4 = (float4*)&lds_sim[(c0 + bcol) * 16 + mbase];
    dst4[0] = make_float4(sv[0], sv[1], sv[2], sv[3]);
    dst4[1] = make_float4(sv[4], sv[5], sv[6], sv[7]);

    __syncthreads();   // sim strip ready

    // Per-row top-17 in registers, lanes 0..15 of wave 0; reads
    // lds_sim[col*16 + row] are bank-conflict-free across lanes.
    if (tid < 16) {
      for (int col = 0; col < 128; ++col) {
        float v = lds_sim[col * 16 + tid];
        if (v > tv[16]) {
          float cv = v; int ci = j0 + col;
#pragma unroll
          for (int s = 0; s < 17; ++s) {
            bool gt = cv > tv[s];
            float nv = gt ? tv[s] : cv;
            int   ni = gt ? ti[s] : ci;
            tv[s] = gt ? cv : tv[s];
            ti[s] = gt ? ci : ti[s];
            cv = nv; ci = ni;
          }
        }
      }
    }

    async_wait0();     // next panel landed in buf^1
    __syncthreads();   // all waves may flip; also fences sim strip reuse
    cur ^= 1;
  }

  if (tid < 16) {  // drop rank 0 (self-similarity) -> knn indices 1..16
    int* o = knn + ((size_t)bb * NPTS + (r0 + tid)) * 16;
#pragma unroll
    for (int s = 1; s < 17; ++s) o[s - 1] = ti[s];
  }

  posd = wave_sum(posd); posc = wave_sum(posc);
  negd = wave_sum(negd); negc = wave_sum(negc);
  if (lane == 0) {
    lds_red[wv * 4 + 0] = posd; lds_red[wv * 4 + 1] = posc;
    lds_red[wv * 4 + 2] = negd; lds_red[wv * 4 + 3] = negc;
  }
  __syncthreads();
  if (tid == 0) {
    float a = 0, b = 0, cc = 0, d = 0;
#pragma unroll
    for (int w = 0; w < 8; ++w) {
      a  += lds_red[w * 4 + 0]; b += lds_red[w * 4 + 1];
      cc += lds_red[w * 4 + 2]; d += lds_red[w * 4 + 3];
    }
    atomicAdd(&lacc[0], a); atomicAdd(&lacc[1], b);
    atomicAdd(&lacc[2], cc); atomicAdd(&lacc[3], d);
  }
}

__global__ void k_fin(const float* __restrict__ lacc, float* __restrict__ out) {
  if (threadIdx.x == 0 && blockIdx.x == 0)
    out[0] = lacc[0] / (lacc[1] + 1e-8f) + lacc[2] / (lacc[3] + 1e-8f);
}

// ---------------------------------------------------------------------------
// Host-side orchestration (stream-ordered, graph-capture safe)
// ---------------------------------------------------------------------------
extern "C" void kernel_launch(void* const* d_in, const int* in_sizes, int n_in,
                              void* d_out, int out_size, void* d_ws, size_t ws_size,
                              hipStream_t stream) {
  (void)in_sizes; (void)n_in; (void)out_size; (void)ws_size;

  const float* xyz   = (const float*)d_in[0];
  const float* few1  = (const float*)d_in[1];
  const float* feb1  = (const float*)d_in[2];
  const float* feg1  = (const float*)d_in[3];
  const float* febe1 = (const float*)d_in[4];
  const float* few2  = (const float*)d_in[5];
  const float* feb2  = (const float*)d_in[6];
  const float* feg2  = (const float*)d_in[7];
  const float* febe2 = (const float*)d_in[8];
  const float* few3  = (const float*)d_in[9];
  const float* feb3  = (const float*)d_in[10];
  const float* snw1  = (const float*)d_in[11];
  const float* snb1  = (const float*)d_in[12];
  const float* sng1  = (const float*)d_in[13];
  const float* snbe1 = (const float*)d_in[14];
  const float* snw2  = (const float*)d_in[15];
  const float* snb2  = (const float*)d_in[16];
  const float* sng2  = (const float*)d_in[17];
  const float* snbe2 = (const float*)d_in[18];
  const float* snw3  = (const float*)d_in[19];
  const float* snb3  = (const float*)d_in[20];

  // workspace layout (floats)
  float* ws    = (float*)d_ws;
  float* stat1 = ws + 0;      // 64
  float* stat2 = ws + 64;     // 128
  float* stat3 = ws + 192;    // 256
  float* stat4 = ws + 448;    // 128
  float* lacc  = ws + 576;    // 4
  float* a1    = ws + 1024;               // 16384*32
  float* a2    = a1 + (size_t)NTOT * 32;  // 16384*64
  float* s1    = a2 + (size_t)NTOT * 64;  // 16384*128
  float* s2    = s1 + (size_t)NTOT * 128; // 16384*64
  unsigned short* featbf = (unsigned short*)(s2 + (size_t)NTOT * 64); // 16384*64 bf16

  // output layout: probs(16384) | loss(1) | knn(4*4096*16 int32) | feat(4*4096*64)
  float* out     = (float*)d_out;
  float* probs   = out;
  float* lossout = out + 16384;
  int*   knn     = (int*)(out + 16385);
  float* feat    = out + 16385 + (size_t)NB * NPTS * 16;

  k_zero<<<4, 256, 0, stream>>>(ws, 1024);
  k_fe1<<<NTOT / 256, 256, 0, stream>>>(xyz, few1, feb1, a1, stat1);
  k_fe2<<<NTOT / 256, 256, 0, stream>>>(a1, stat1, feg1, febe1, few2, feb2, a2, stat2);
  k_fe3<<<NTOT / 256, 256, 0, stream>>>(a2, stat2, feg2, febe2, few3, feb3, feat);
  k_sn1<<<NTOT / 256, 256, 0, stream>>>(feat, featbf, xyz, snw1, snb1, s1, stat3);
  k_sn2<<<NTOT / 256, 256, 0, stream>>>(s1, stat3, sng1, snbe1, snw2, snb2, s2, stat4);
  k_sn3<<<NTOT / 256, 256, 0, stream>>>(s2, stat4, sng2, snbe2, snw3, snb3, probs);

  dim3 grid(NPTS / 16, NB);
  k_sim<<<grid, 256, 0, stream>>>(featbf, probs, knn, lacc);
  k_fin<<<1, 32, 0, stream>>>(lacc, lossout);
}
